// MultiHeadAttention_55843164783189
// MI455X (gfx1250) — matrix operand order repaired and checked by reading, not compiled, not run on hardware
//
#include <hip/hip_runtime.h>
#include <stdint.h>

// ---------------------------------------------------------------- types
typedef _Float16 v8h  __attribute__((ext_vector_type(8)));
typedef _Float16 v16h __attribute__((ext_vector_type(16)));
typedef float    v8f  __attribute__((ext_vector_type(8)));
typedef unsigned int u32x4 __attribute__((ext_vector_type(4)));
typedef int          i32x4 __attribute__((ext_vector_type(4)));
typedef int          i32x8 __attribute__((ext_vector_type(8)));

#define DEVINL __device__ __forceinline__

static constexpr int Bc  = 4;
static constexpr int Sc  = 1024;
static constexpr int Dc  = 1024;
static constexpr int Hc  = 16;
static constexpr int DKc = 64;
static constexpr float NEG_INF = -1000000000.0f;

#if defined(__HIP_DEVICE_COMPILE__) && __has_builtin(__builtin_amdgcn_tensor_load_to_lds)
#define HAS_TDM 1
#else
#define HAS_TDM 0
#endif

DEVINL v16h cat8(v8h a, v8h b) {
  return __builtin_shufflevector(a, b, 0,1,2,3,4,5,6,7,8,9,10,11,12,13,14,15);
}

// A-matrix 16x32 f16 fragment (CDNA5 layout):
//   lanes 0-15 : M=lane,    K chunks {0..7, 16..23}
//   lanes 16-31: M=lane-16, K chunks {8..15, 24..31}
DEVINL v16h load_A16x32(const _Float16* base, int pitch, int lane) {
  const int half = lane >> 4;
  const int m    = lane & 15;
  const _Float16* row = base + m * pitch;
  v8h lo = *(const v8h*)(row + half * 8);
  v8h hi = *(const v8h*)(row + 16 + half * 8);
  return cat8(lo, hi);
}

// B-matrix 32x16 f16 fragment: lane l = column N=l%16, K chunks as A layout.
DEVINL v16h load_B32x16(const _Float16* base, int pitch, int lane) {
  const int half = lane >> 4;
  const _Float16* col = base + (lane & 15);
  v16h r;
#pragma unroll
  for (int s = 0; s < 8; ++s) {
    r[s]     = col[(8 * half + s) * pitch];
    r[s + 8] = col[(16 + 8 * half + s) * pitch];
  }
  return r;
}

DEVINL v8f wmma_f16(v16h a, v16h b, v8f c) {
  return __builtin_amdgcn_wmma_f32_16x16x32_f16(false, a, false, b, (short)0, c,
                                                false, false);
}

// ---- TDM helpers -------------------------------------------------------
// 2-D f16 tile load: 64 rows x 64 elems, row stride `stride_elems`, into LDS
// at byte offset `lds_off` with pitch 72 f16 (pad 4 DWORDs after each
// 32-DWORD row: pad_interval code 4, pad_amount code 3).
DEVINL void tdm_load_tile64(uint32_t lds_off, const void* gptr,
                            uint64_t stride_elems) {
#if HAS_TDM
  const uint64_t ga = (uint64_t)(uintptr_t)gptr;
  u32x4 g0 = {};
  g0.x = 1u;                                                 // count = 1
  g0.y = lds_off;                                            // lds_addr
  g0.z = (uint32_t)ga;                                       // global_addr[31:0]
  g0.w = (uint32_t)((ga >> 32) & 0x1FFFFFFu) | (2u << 30);   // ga[56:32] | type=2
  i32x8 g1 = {};
  // data_size=1 (2B) | pad_enable | pad_interval=4 (32 DW) | pad_amount=3 (4 DW)
  g1[0] = (1 << 16) | (1 << 20) | (4 << 22) | (3 << 25);
  g1[1] = (int)(1024u << 16);            // tensor_dim0[15:0] = 1024 (bits 79:48)
  g1[2] = (int)(64u << 16);              // tensor_dim1[15:0] = 64   (bits 111:80)
  g1[3] = (int)(64u << 16);              // tile_dim0 = 64           (bits 127:112)
  g1[4] = (int)(64u);                    // tile_dim1 = 64, tile_dim2 = 0
  g1[5] = (int)(uint32_t)stride_elems;   // tensor_dim0_stride[31:0]
  g1[6] = (int)(uint32_t)(stride_elems >> 32);
  g1[7] = 0;
  i32x4 z4 = {};
#if __clang_major__ >= 23
  i32x8 z8 = {};
  __builtin_amdgcn_tensor_load_to_lds(g0, g1, z4, z4, z8, 0);
#else
  __builtin_amdgcn_tensor_load_to_lds(g0, g1, z4, z4, 0);
#endif
#else
  (void)lds_off; (void)gptr; (void)stride_elems;
#endif
}

DEVINL void wait_tensorcnt_le2() {
#if defined(__HIP_DEVICE_COMPILE__)
#if __has_builtin(__builtin_amdgcn_s_wait_tensorcnt)
  __builtin_amdgcn_s_wait_tensorcnt(2);
#else
  asm volatile("s_wait_tensorcnt 0x2" ::: "memory");
#endif
#endif
}
DEVINL void wait_tensorcnt_le0() {
#if defined(__HIP_DEVICE_COMPILE__)
#if __has_builtin(__builtin_amdgcn_s_wait_tensorcnt)
  __builtin_amdgcn_s_wait_tensorcnt(0);
#else
  asm volatile("s_wait_tensorcnt 0x0" ::: "memory");
#endif
#endif
}

// ---------------------------------------------------------------- GEMM
// C[r][c] = sum_k X[r][k] * W[c][k] + bias[c]
// MODE 0: X f32, out f16 head-split rowmajor  qh[b][h][s][dk]
// MODE 1: X f32, out f16 head-split TRANSPOSE kt[b][h][dk][s]
// MODE 2: X f32, out f16 head-split rowmajor  vh[b][h][s][dk]
// MODE 3: X f16, out f32 plain rowmajor       out[r][c]
template <int MODE>
__global__ __launch_bounds__(256) void gemm_proj(
    const float* __restrict__ Xf, const _Float16* __restrict__ Xh,
    const float* __restrict__ W, const float* __restrict__ bias,
    _Float16* __restrict__ outh, float* __restrict__ outf) {
  __shared__ _Float16 Xl[128][40];   // rowmajor [m][k], 80B pitch (16B aligned)
  __shared__ _Float16 Wl[32][136];   // transposed [k][n], 272B pitch

  const int lane = threadIdx.x & 31;
  const int wave = threadIdx.x >> 5;
  const int tm = blockIdx.y * 128;
  const int tn = blockIdx.x * 128;
  const int wm = (wave & 3) * 32;    // 4 waves along M
  const int wn = (wave >> 2) * 64;   // 2 waves along N

  v8f zero = {};
  v8f acc[2][4];
#pragma unroll
  for (int i = 0; i < 2; ++i)
#pragma unroll
    for (int j = 0; j < 4; ++j) acc[i][j] = zero;

  const int t  = threadIdx.x;
  const int xr = t >> 1;             // 0..127
  const int xk = (t & 1) * 16;       // 0 or 16

  for (int k0 = 0; k0 < Dc; k0 += 32) {
    // ---- stage X tile (128 x 32) ----
    if (MODE == 3) {
      const _Float16* src = Xh + (size_t)(tm + xr) * Dc + k0 + xk;
      *(v8h*)&Xl[xr][xk]     = *(const v8h*)(src);
      *(v8h*)&Xl[xr][xk + 8] = *(const v8h*)(src + 8);
    } else {
      const float* src = Xf + (size_t)(tm + xr) * Dc + k0 + xk;
#pragma unroll
      for (int j = 0; j < 16; ++j) Xl[xr][xk + j] = (_Float16)src[j];
    }
    // ---- stage W tile transposed (32 x 128) ----
    {
      const float* src = W + (size_t)(tn + xr) * Dc + k0 + xk;
#pragma unroll
      for (int j = 0; j < 16; ++j) Wl[xk + j][xr] = (_Float16)src[j];
    }
    __syncthreads();

    v16h Afr[2];
#pragma unroll
    for (int m16 = 0; m16 < 2; ++m16)
      Afr[m16] = load_A16x32(&Xl[wm + m16 * 16][0], 40, lane);
#pragma unroll
    for (int nt = 0; nt < 4; ++nt) {
      v16h Bfr = load_B32x16(&Wl[0][wn + nt * 16], 136, lane);
#pragma unroll
      for (int m16 = 0; m16 < 2; ++m16)
        acc[m16][nt] = wmma_f16(Afr[m16], Bfr, acc[m16][nt]);
    }
    __syncthreads();
  }

  // ---- epilogue: bias + scatter ----
  const int half = lane >> 4;
  const int nl   = lane & 15;
#pragma unroll
  for (int m16 = 0; m16 < 2; ++m16) {
#pragma unroll
    for (int nt = 0; nt < 4; ++nt) {
#pragma unroll
      for (int vr = 0; vr < 8; ++vr) {
        const int R = tm + wm + m16 * 16 + vr + 8 * half;  // global row (b*S+s)
        const int C = tn + wn + nt * 16 + nl;              // global col (h*64+dk)
        float v = acc[m16][nt][vr] + bias[C];
        if (MODE == 0 || MODE == 2) {
          const int b = R >> 10, s = R & 1023, h = C >> 6, dk = C & 63;
          outh[(((size_t)b * Hc + h) * Sc + s) * DKc + dk] = (_Float16)v;
        } else if (MODE == 1) {
          const int b = R >> 10, s = R & 1023, h = C >> 6, dk = C & 63;
          outh[(((size_t)b * Hc + h) * DKc + dk) * Sc + s] = (_Float16)v;
        } else {
          outf[(size_t)R * Dc + C] = v;
        }
      }
    }
  }
}

// ---------------------------------------------------------------- attention
// One block = 8 waves = 128 query rows of one (b,h). KV tiles of 64.
// K/V tiles double-buffered in LDS, staged by the Tensor Data Mover.
__global__ __launch_bounds__(256) void attn_kernel(
    const _Float16* __restrict__ qh, const _Float16* __restrict__ kt,
    const _Float16* __restrict__ vh, const float* __restrict__ token,
    const float* __restrict__ statement, const float* __restrict__ dflow,
    const float* __restrict__ cflow, const int* __restrict__ mask,
    _Float16* __restrict__ ctx) {
  __shared__ _Float16 Klds[2][64][72];    // [dk][kv]  (B-operand rows for QK^T)
  __shared__ _Float16 Vlds[2][64][72];    // [kv][dk]  (B-operand rows for P V)
  __shared__ _Float16 Plds[8][16][72];    // per-wave P staging, A-layout source

  const int lane = threadIdx.x & 31;
  const int wave = threadIdx.x >> 5;
  const int qt = blockIdx.x, h = blockIdx.y, b = blockIdx.z;
  const int bh = b * Hc + h;
  const int q0 = qt * 128 + wave * 16;

  // head group: [4 token][4 statement][2 data_flow][2 control_flow][4 std]
  const float* adj = nullptr;
  int cls = 0;  // 0=none, 1=local(-1e9), 2=enhance(+5*adj*|s|)
  if (h < 4)       { adj = token;     cls = 1; }
  else if (h < 8)  { adj = statement; cls = 1; }
  else if (h < 10) { adj = dflow;     cls = 2; }
  else if (h < 12) { adj = cflow;     cls = 2; }
  const float* adjbase = adj ? adj + (size_t)b * Sc * Sc : nullptr;
  const int*   mbase   = mask + (size_t)b * Sc * Sc;

  // ---- preload Q fragments (16 rows x 64 dk -> two K=32 A tiles) ----
  const _Float16* qbase = qh + ((size_t)bh * Sc + q0) * DKc;
  v16h Afr[2];
  {
    const int hf = lane >> 4, m = lane & 15;
    const _Float16* row = qbase + (size_t)m * DKc;
#pragma unroll
    for (int kc = 0; kc < 2; ++kc) {
      v8h lo = *(const v8h*)(row + kc * 32 + hf * 8);
      v8h hi = *(const v8h*)(row + kc * 32 + 16 + hf * 8);
      Afr[kc] = cat8(lo, hi);
    }
  }

  v8f zero = {};
  v8f O[4];
#pragma unroll
  for (int dt = 0; dt < 4; ++dt) O[dt] = zero;
  float mrow[8], lrow[8];
#pragma unroll
  for (int r = 0; r < 8; ++r) { mrow[r] = -3.0e38f; lrow[r] = 0.0f; }

  const _Float16* ktbase = kt + (size_t)bh * DKc * Sc;
  const _Float16* vbase  = vh + (size_t)bh * Sc * DKc;

  const int half = lane >> 4;
  const int nl   = lane & 15;

#if HAS_TDM
  uint32_t koff[2], voff[2];
  koff[0] = (uint32_t)(uintptr_t)&Klds[0][0][0];
  koff[1] = (uint32_t)(uintptr_t)&Klds[1][0][0];
  voff[0] = (uint32_t)(uintptr_t)&Vlds[0][0][0];
  voff[1] = (uint32_t)(uintptr_t)&Vlds[1][0][0];
  // prologue: DMA tile 0 into buffer 0
  if (wave == 0) {
    tdm_load_tile64(koff[0], ktbase, (uint64_t)Sc);
    tdm_load_tile64(voff[0], vbase, (uint64_t)DKc);
  }
#endif

  for (int it = 0; it < Sc / 64; ++it) {
    const int kv0 = it * 64;
#if HAS_TDM
    const int buf = it & 1;
    __syncthreads();  // all waves done reading buf^1 (iteration it-1)
    if (wave == 0) {
      if (it + 1 < Sc / 64) {
        // DMA next tile while everyone computes this one
        tdm_load_tile64(koff[buf ^ 1], ktbase + (kv0 + 64), (uint64_t)Sc);
        tdm_load_tile64(voff[buf ^ 1], vbase + (size_t)(kv0 + 64) * DKc,
                        (uint64_t)DKc);
        wait_tensorcnt_le2();  // current tile done (in-order), next in flight
      } else {
        wait_tensorcnt_le0();
      }
    }
    __syncthreads();  // publish tile `it` to all waves
#else
    const int buf = 0;
    {
      const int r = threadIdx.x >> 2;
      const int cchunk = (threadIdx.x & 3) * 16;
      const _Float16* ksrc = ktbase + (size_t)r * Sc + kv0 + cchunk;
      *(v8h*)&Klds[0][r][cchunk]     = *(const v8h*)(ksrc);
      *(v8h*)&Klds[0][r][cchunk + 8] = *(const v8h*)(ksrc + 8);
      const _Float16* vsrc = vbase + (size_t)(kv0 + r) * DKc + cchunk;
      *(v8h*)&Vlds[0][r][cchunk]     = *(const v8h*)(vsrc);
      *(v8h*)&Vlds[0][r][cchunk + 8] = *(const v8h*)(vsrc + 8);
    }
    __syncthreads();
#endif

    // ---- scores: 16q x 64kv via WMMA, K-chained over dk ----
    v8f sc[4];
#pragma unroll
    for (int nt = 0; nt < 4; ++nt) {
      v8f c = zero;
#pragma unroll
      for (int kc = 0; kc < 2; ++kc) {
        v16h Bfr = load_B32x16(&Klds[buf][kc * 32][nt * 16], 72, lane);
        c = wmma_f16(Afr[kc], Bfr, c);
      }
      sc[nt] = c;
    }

    // ---- prefetch next KV tile's mask/adjacency cachelines ----
    if (kv0 + 64 < Sc) {
#pragma unroll
      for (int nt = 0; nt < 4; ++nt) {
        const size_t pidx = (size_t)(q0 + 8 * half) * Sc + kv0 + 64 + nt * 16 + nl;
        if (cls) __builtin_prefetch(adjbase + pidx, 0, 1);
        __builtin_prefetch(mbase + pidx, 0, 1);
      }
    }

    // ---- fused epilogue: scale, local mask / enhance, attention mask ----
#pragma unroll
    for (int nt = 0; nt < 4; ++nt) {
      const int col = kv0 + nt * 16 + nl;
#pragma unroll
      for (int vr = 0; vr < 8; ++vr) {
        const int row = q0 + vr + 8 * half;
        const size_t idx = (size_t)row * Sc + col;
        float s = sc[nt][vr] * 0.125f;  // 1/sqrt(64)
        if (cls == 1)      s += NEG_INF * adjbase[idx];
        else if (cls == 2) s += 5.0f * adjbase[idx] * fabsf(s);
        if (mbase[idx] == 0) s = NEG_INF;
        sc[nt][vr] = s;
      }
    }

    // ---- online softmax (row r lives in VGPR r across a 16-lane half) ----
#pragma unroll
    for (int vr = 0; vr < 8; ++vr) {
      float rm = fmaxf(fmaxf(sc[0][vr], sc[1][vr]), fmaxf(sc[2][vr], sc[3][vr]));
      rm = fmaxf(rm, __shfl_xor(rm, 1, 16));
      rm = fmaxf(rm, __shfl_xor(rm, 2, 16));
      rm = fmaxf(rm, __shfl_xor(rm, 4, 16));
      rm = fmaxf(rm, __shfl_xor(rm, 8, 16));
      const float mnew  = fmaxf(mrow[vr], rm);
      const float alpha = __expf(mrow[vr] - mnew);
      float psum = 0.0f;
#pragma unroll
      for (int nt = 0; nt < 4; ++nt) {
        float p = __expf(sc[nt][vr] - mnew);
        sc[nt][vr] = p;
        psum += p;
      }
      psum += __shfl_xor(psum, 1, 16);
      psum += __shfl_xor(psum, 2, 16);
      psum += __shfl_xor(psum, 4, 16);
      psum += __shfl_xor(psum, 8, 16);
      lrow[vr] = lrow[vr] * alpha + psum;
      mrow[vr] = mnew;
#pragma unroll
      for (int dt = 0; dt < 4; ++dt) O[dt][vr] *= alpha;
    }

    // ---- stage P (f16) into per-wave LDS for A-operand reload ----
#pragma unroll
    for (int nt = 0; nt < 4; ++nt)
#pragma unroll
      for (int vr = 0; vr < 8; ++vr)
        Plds[wave][vr + 8 * half][nt * 16 + nl] = (_Float16)sc[nt][vr];

    // ---- O += P @ V ----
#pragma unroll
    for (int kc = 0; kc < 2; ++kc) {
      v16h Pa = load_A16x32(&Plds[wave][0][kc * 32], 72, lane);
#pragma unroll
      for (int dt = 0; dt < 4; ++dt) {
        v16h Bv = load_B32x16(&Vlds[buf][kc * 32][dt * 16], 72, lane);
        O[dt] = wmma_f16(Pa, Bv, O[dt]);
      }
    }
#if !HAS_TDM
    __syncthreads();
#endif
  }

  // ---- normalize and store ctx as [b][s][h*64+dk] f16 ----
#pragma unroll
  for (int vr = 0; vr < 8; ++vr) {
    const float inv = 1.0f / lrow[vr];
    const int row = q0 + vr + 8 * half;
#pragma unroll
    for (int dt = 0; dt < 4; ++dt) {
      float o = O[dt][vr] * inv;
      ctx[((size_t)b * Sc + row) * Dc + h * DKc + dt * 16 + nl] = (_Float16)o;
    }
  }
}

// ---------------------------------------------------------------- launch
extern "C" void kernel_launch(void* const* d_in, const int* in_sizes, int n_in,
                              void* d_out, int out_size, void* d_ws,
                              size_t ws_size, hipStream_t stream) {
  (void)in_sizes; (void)n_in; (void)out_size; (void)ws_size;
  const float* Q    = (const float*)d_in[0];
  const float* K    = (const float*)d_in[1];
  const float* V    = (const float*)d_in[2];
  const float* tok  = (const float*)d_in[3];
  const float* stmt = (const float*)d_in[4];
  const float* dfl  = (const float*)d_in[5];
  const float* cfl  = (const float*)d_in[6];
  // d_in[7] = zero_matrix (unused: contributes exactly zero)
  const int*   msk  = (const int*)d_in[8];
  const float* Wq = (const float*)d_in[9];
  const float* bq = (const float*)d_in[10];
  const float* Wk = (const float*)d_in[11];
  const float* bk = (const float*)d_in[12];
  const float* Wv = (const float*)d_in[13];
  const float* bv = (const float*)d_in[14];
  const float* Wo = (const float*)d_in[15];
  const float* bo = (const float*)d_in[16];
  float* out = (float*)d_out;

  _Float16* ws = (_Float16*)d_ws;
  const size_t seg = (size_t)Bc * Hc * Sc * DKc;  // 4M f16 elems per tensor
  _Float16* qh  = ws;
  _Float16* kt  = ws + seg;
  _Float16* vh  = ws + 2 * seg;
  _Float16* ctx = ws + 3 * seg;

  dim3 blk(256);
  dim3 gg(Dc / 128, (Bc * Sc) / 128);  // 8 x 32 tiles
  gemm_proj<0><<<gg, blk, 0, stream>>>(Q, nullptr, Wq, bq, qh, nullptr);
  gemm_proj<1><<<gg, blk, 0, stream>>>(K, nullptr, Wk, bk, kt, nullptr);
  gemm_proj<2><<<gg, blk, 0, stream>>>(V, nullptr, Wv, bv, vh, nullptr);

  dim3 ga(Sc / 128, Hc, Bc);  // 8 x 16 x 4
  attn_kernel<<<ga, blk, 0, stream>>>(qh, kt, vh, tok, stmt, dfl, cfl, msk, ctx);

  gemm_proj<3><<<gg, blk, 0, stream>>>(nullptr, ctx, Wo, bo, nullptr, out);
}
